// ERAFT_7095285973763
// MI455X (gfx1250) — compile-verified, matrix operand order checked by reading, not run on hardware
//
#include <hip/hip_runtime.h>
#include <hip/hip_bf16.h>
#include <math.h>

// ---------------- problem constants (from reference) ----------------
#define PX   4800      // 60*80 pixels
#define HH   60
#define WW   80
#define ITERS 12
#define NTHREADS 256

typedef __attribute__((ext_vector_type(16))) _Float16 v16h;
typedef __attribute__((ext_vector_type(8)))  _Float16 v8h;
typedef __attribute__((ext_vector_type(8)))  float    v8f;

static __device__ __forceinline__ float dev_sigmoid(float v) {
    return 1.0f / (1.0f + __expf(-v));
}

// =====================================================================
// WMMA GEMM, 16x64 C-strip per wave (4 accumulators, A-fragment reuse):
//   C[Mpad x N] = act(alpha * A[Mpad x Kpad] * B + bias)
//   A  : Mpad x Kpad row-major f16 (weights / f1t, zero padded)
//   BT : N x Kpad row-major f16 (B^T, K-contiguous per pixel)
//   out: f32 (Cf) or f16 (Ch) row-major Mpad x N
// Per-lane fragment layout per CDNA5 ISA (16-bit A/B striping):
//   lanes 0-15  : K = kk+0..7  and kk+16..23  (two b128 loads)
//   lanes 16-31 : K = kk+8..15 and kk+24..31
// 4 independent accumulator chains hide WMMA->WMMA RAW latency.
// =====================================================================
__global__ void k_gemm_wmma(const _Float16* __restrict__ A,
                            const _Float16* __restrict__ BT,
                            float* __restrict__ Cf,
                            _Float16* __restrict__ Ch,
                            const float* __restrict__ bias,
                            int Mpad, int N, int Kpad,
                            float alpha, int op, int outHalf)
{
    const int wave = blockIdx.x * (blockDim.x >> 5) + (threadIdx.x >> 5);
    const int lane = threadIdx.x & 31;
    const int ntiles = N >> 6;                  // 64-wide N strips
    const int tiles  = (Mpad >> 4) * ntiles;
    if (wave >= tiles) return;

    const int mt = wave / ntiles;
    const int nt = wave % ntiles;
    const int m0 = mt << 4;
    const int n0 = nt << 6;
    const int lm = lane & 15;
    const int hi = lane >> 4;                   // half-wave selector

    const _Float16* pA  = A  + (size_t)(m0 + lm) * Kpad + (hi ? 8 : 0);
    const _Float16* pB0 = BT + (size_t)(n0 + lm) * Kpad + (hi ? 8 : 0);
    const size_t bstride = (size_t)16 * Kpad;   // next 16-col B tile

    v8f acc0 = {}, acc1 = {}, acc2 = {}, acc3 = {};
    for (int kk = 0; kk < Kpad; kk += 32) {
        __builtin_prefetch(pA  + kk + 128, 0, 1);   // -> global_prefetch_b8
        __builtin_prefetch(pB0 + kk + 128, 0, 1);
        v8h a0 = *(const v8h*)(pA + kk);
        v8h a1 = *(const v8h*)(pA + kk + 16);
        v16h a = __builtin_shufflevector(a0, a1, 0,1,2,3,4,5,6,7,8,9,10,11,12,13,14,15);

        const _Float16* pB = pB0 + kk;
        v8h b00 = *(const v8h*)(pB);
        v8h b01 = *(const v8h*)(pB + 16);
        v8h b10 = *(const v8h*)(pB + bstride);
        v8h b11 = *(const v8h*)(pB + bstride + 16);
        v8h b20 = *(const v8h*)(pB + 2 * bstride);
        v8h b21 = *(const v8h*)(pB + 2 * bstride + 16);
        v8h b30 = *(const v8h*)(pB + 3 * bstride);
        v8h b31 = *(const v8h*)(pB + 3 * bstride + 16);
        v16h b0 = __builtin_shufflevector(b00, b01, 0,1,2,3,4,5,6,7,8,9,10,11,12,13,14,15);
        v16h b1 = __builtin_shufflevector(b10, b11, 0,1,2,3,4,5,6,7,8,9,10,11,12,13,14,15);
        v16h b2 = __builtin_shufflevector(b20, b21, 0,1,2,3,4,5,6,7,8,9,10,11,12,13,14,15);
        v16h b3 = __builtin_shufflevector(b30, b31, 0,1,2,3,4,5,6,7,8,9,10,11,12,13,14,15);

        acc0 = __builtin_amdgcn_wmma_f32_16x16x32_f16(false, a, false, b0, (short)0, acc0, false, false);
        acc1 = __builtin_amdgcn_wmma_f32_16x16x32_f16(false, a, false, b1, (short)0, acc1, false, false);
        acc2 = __builtin_amdgcn_wmma_f32_16x16x32_f16(false, a, false, b2, (short)0, acc2, false, false);
        acc3 = __builtin_amdgcn_wmma_f32_16x16x32_f16(false, a, false, b3, (short)0, acc3, false, false);
    }

    // C/D layout: VGPR r -> row m0 + r + 8*hi, col (n0 + 16*t) + (lane&15)
    v8f accs[4] = {acc0, acc1, acc2, acc3};
    #pragma unroll
    for (int t = 0; t < 4; ++t) {
        #pragma unroll
        for (int r = 0; r < 8; ++r) {
            int m = m0 + r + 8 * hi;
            float v = accs[t][r] * alpha;
            if (bias) v += bias[m];
            if      (op == 1) v = fmaxf(v, 0.0f);
            else if (op == 2) v = dev_sigmoid(v);
            else if (op == 3) v = tanhf(v);
            size_t idx = (size_t)m * N + n0 + 16 * t + lm;
            if (outHalf) Ch[idx] = (_Float16)v;
            else         Cf[idx] = v;
        }
    }
}

// ------------- f32 (rows x cols) -> f16 transpose: dst[j*rows+i] -------------
__global__ void k_f32_to_h_t(_Float16* __restrict__ dst, const float* __restrict__ src,
                             int rows, int cols)
{
    size_t t = (size_t)blockIdx.x * blockDim.x + threadIdx.x;
    size_t total = (size_t)rows * cols;
    if (t >= total) return;
    int j = (int)(t % cols);
    int i = (int)(t / cols);
    dst[(size_t)j * rows + i] = (_Float16)src[t];
}

// ------------- im2col, transposed output: dst[p*Kpad + k] (f16) -------------
__global__ void k_im2col_t(_Float16* __restrict__ dst, const float* __restrict__ src,
                           int Cin, int kh, int kw, int ph, int pw, int Kpad)
{
    size_t t = (size_t)blockIdx.x * blockDim.x + threadIdx.x;
    size_t total = (size_t)Kpad * PX;
    if (t >= total) return;
    int k = (int)(t % Kpad);
    int p = (int)(t / Kpad);
    int K = Cin * kh * kw;
    float v = 0.0f;
    if (k < K) {
        int j = k % kw;
        int i = (k / kw) % kh;
        int c = k / (kw * kh);
        int x = p % WW + j - pw;
        int y = p / WW + i - ph;
        if (x >= 0 && x < WW && y >= 0 && y < HH)
            v = src[(size_t)c * PX + y * WW + x];
    }
    dst[(size_t)p * Kpad + k] = (_Float16)v;
}

// ------------- weight pack: (Cout x K) f32 -> (Mpad x Kpad) f16, zero pad ----
__global__ void k_pack_w(_Float16* __restrict__ dst, const float* __restrict__ src,
                         int Cout, int K, int Mpad, int Kpad)
{
    size_t t = (size_t)blockIdx.x * blockDim.x + threadIdx.x;
    size_t total = (size_t)Mpad * Kpad;
    if (t >= total) return;
    int k = (int)(t % Kpad);
    int m = (int)(t / Kpad);
    float v = (m < Cout && k < K) ? src[(size_t)m * K + k] : 0.0f;
    dst[t] = (_Float16)v;
}

__global__ void k_pack_b(float* __restrict__ dst, const float* __restrict__ src,
                         int Cout, int Mpad, float scale)
{
    int t = blockIdx.x * blockDim.x + threadIdx.x;
    if (t >= Mpad) return;
    dst[t] = (t < Cout) ? src[t] * scale : 0.0f;
}

// ------------- 2x2 average pool on f16 pyramid level -------------
__global__ void k_avgpool2(_Float16* __restrict__ dst, const _Float16* __restrict__ src,
                           int hh, int ww)
{
    int oh = hh >> 1, ow = ww >> 1;
    size_t t = (size_t)blockIdx.x * blockDim.x + threadIdx.x;
    size_t total = (size_t)PX * oh * ow;
    if (t >= total) return;
    int x = (int)(t % ow);
    int y = (int)((t / ow) % oh);
    int n = (int)(t / ((size_t)ow * oh));
    const _Float16* s = src + ((size_t)n * hh + 2 * y) * ww + 2 * x;
    float v = ((float)s[0] + (float)s[1] + (float)s[ww] + (float)s[ww + 1]) * 0.25f;
    dst[t] = (_Float16)v;
}

// ------------- correlation pyramid bilinear lookup -> feat (324 x PX) f32 ----
__global__ void k_corr_lookup(float* __restrict__ feat,
                              const _Float16* __restrict__ p0,
                              const _Float16* __restrict__ p1,
                              const _Float16* __restrict__ p2,
                              const _Float16* __restrict__ p3,
                              const float* __restrict__ cx,
                              const float* __restrict__ cy)
{
    size_t t = (size_t)blockIdx.x * blockDim.x + threadIdx.x;
    size_t total = (size_t)324 * PX;
    if (t >= total) return;
    int p   = (int)(t % PX);
    int ch  = (int)(t / PX);
    int lvl = ch / 81;
    int tap = ch % 81;
    float d0 = (float)(tap / 9 - 4);   // delta[...,0] (added to x in reference)
    float d1 = (float)(tap % 9 - 4);   // delta[...,1] (added to y)

    const _Float16* img; int hh, ww;
    if      (lvl == 0) { img = p0; hh = 60; ww = 80; }
    else if (lvl == 1) { img = p1; hh = 30; ww = 40; }
    else if (lvl == 2) { img = p2; hh = 15; ww = 20; }
    else               { img = p3; hh = 7;  ww = 10; }

    float sc = 1.0f / (float)(1 << lvl);
    float x = cx[p] * sc + d0;
    float y = cy[p] * sc + d1;

    float x0f = floorf(x), y0f = floorf(y);
    float wx = x - x0f, wy = y - y0f;
    int x0 = (int)x0f, y0 = (int)y0f;
    const _Float16* base = img + (size_t)p * hh * ww;

    float acc = 0.0f;
    #pragma unroll
    for (int q = 0; q < 4; ++q) {
        int ix = x0 + (q & 1);
        int iy = y0 + (q >> 1);
        float wt = ((q & 1) ? wx : 1.0f - wx) * ((q >> 1) ? wy : 1.0f - wy);
        if (ix >= 0 && ix < ww && iy >= 0 && iy < hh)
            acc += (float)base[iy * ww + ix] * wt;
    }
    feat[(size_t)ch * PX + p] = acc;
}

// ------------- misc elementwise -------------
__global__ void k_split_cnet(float* __restrict__ net, float* __restrict__ hx,
                             const float* __restrict__ cnet)
{
    size_t t = (size_t)blockIdx.x * blockDim.x + threadIdx.x;
    if (t >= (size_t)256 * PX) return;
    int c = (int)(t / PX);
    float v = cnet[t];
    if (c < 128) net[t] = tanhf(v);
    else         hx[t]  = fmaxf(v, 0.0f);   // inp lives at hx rows 128..255
}

__global__ void k_copy_f32(float* __restrict__ dst, const float* __restrict__ src, size_t n)
{
    size_t t = (size_t)blockIdx.x * blockDim.x + threadIdx.x;
    if (t < n) dst[t] = src[t];
}

__global__ void k_mul_f32(float* __restrict__ dst, const float* __restrict__ a,
                          const float* __restrict__ b, size_t n)
{
    size_t t = (size_t)blockIdx.x * blockDim.x + threadIdx.x;
    if (t < n) dst[t] = a[t] * b[t];
}

__global__ void k_gru(float* __restrict__ net, const float* __restrict__ z,
                      const float* __restrict__ q, size_t n)
{
    size_t t = (size_t)blockIdx.x * blockDim.x + threadIdx.x;
    if (t < n) {
        float zz = z[t];
        net[t] = (1.0f - zz) * net[t] + zz * q[t];
    }
}

__global__ void k_coords_init(float* __restrict__ cx, float* __restrict__ cy)
{
    int t = blockIdx.x * blockDim.x + threadIdx.x;
    if (t >= PX) return;
    cx[t] = (float)(t % WW);
    cy[t] = (float)(t / WW);
}

__global__ void k_flow_from_coords(float* __restrict__ flow,
                                   const float* __restrict__ cx,
                                   const float* __restrict__ cy)
{
    int t = blockIdx.x * blockDim.x + threadIdx.x;
    if (t >= PX) return;
    flow[t]      = cx[t] - (float)(t % WW);
    flow[PX + t] = cy[t] - (float)(t / WW);
}

__global__ void k_add_coords(float* __restrict__ cx, float* __restrict__ cy,
                             const float* __restrict__ df)
{
    int t = blockIdx.x * blockDim.x + threadIdx.x;
    if (t >= PX) return;
    cx[t] += df[t];
    cy[t] += df[PX + t];
}

// ------------- convex upsample: out 2 x 480 x 640 per iteration -------------
__global__ void k_upsample(float* __restrict__ out, const float* __restrict__ mask,
                           const float* __restrict__ flow)
{
    size_t t = (size_t)blockIdx.x * blockDim.x + threadIdx.x;
    size_t total = (size_t)2 * PX * 64;
    if (t >= total) return;
    int j  = (int)(t & 7);
    int i  = (int)((t >> 3) & 7);
    int pc = (int)(t >> 6);
    int p  = pc % PX;
    int c  = pc / PX;
    int h = p / WW, w = p % WW;

    float m[9], mx = -1e30f;
    #pragma unroll
    for (int k = 0; k < 9; ++k) {
        m[k] = mask[((size_t)(k * 64 + i * 8 + j)) * PX + p];
        mx = fmaxf(mx, m[k]);
    }
    float s = 0.0f;
    #pragma unroll
    for (int k = 0; k < 9; ++k) { m[k] = __expf(m[k] - mx); s += m[k]; }
    float inv = 1.0f / s;

    const float* f = flow + (size_t)c * PX;
    float acc = 0.0f;
    #pragma unroll
    for (int k = 0; k < 9; ++k) {
        int yy = h + k / 3 - 1;
        int xx = w + k % 3 - 1;
        float fv = (yy >= 0 && yy < HH && xx >= 0 && xx < WW) ? 8.0f * f[yy * WW + xx] : 0.0f;
        acc += m[k] * inv * fv;
    }
    out[(size_t)c * 307200 + (size_t)(h * 8 + i) * 640 + (w * 8 + j)] = acc;
}

// =====================================================================
// Host orchestration
// =====================================================================
static inline unsigned nb(size_t n) { return (unsigned)((n + NTHREADS - 1) / NTHREADS); }
static inline int rup(int v, int a) { return (v + a - 1) / a * a; }

extern "C" void kernel_launch(void* const* d_in, const int* in_sizes, int n_in,
                              void* d_out, int out_size, void* d_ws, size_t ws_size,
                              hipStream_t stream)
{
    const float* fmap1 = (const float*)d_in[0];
    const float* fmap2 = (const float*)d_in[1];
    const float* cnet  = (const float*)d_in[2];
    float* out = (float*)d_out;

    // ---- workspace carve-out ----
    char* ws = (char*)d_ws;
    size_t off = 0;
    auto alloc = [&](size_t bytes) -> char* {
        char* p = ws + off;
        off = (off + bytes + 255) & ~(size_t)255;
        return p;
    };

    _Float16* f1t  = (_Float16*)alloc((size_t)PX * 256 * 2);
    _Float16* f2t  = (_Float16*)alloc((size_t)PX * 256 * 2);
    _Float16* pyr0 = (_Float16*)alloc((size_t)PX * 4800 * 2);
    _Float16* pyr1 = (_Float16*)alloc((size_t)PX * 1200 * 2);
    _Float16* pyr2 = (_Float16*)alloc((size_t)PX * 300  * 2);
    _Float16* pyr3 = (_Float16*)alloc((size_t)PX * 70   * 2);

    // layers: {weight input idx, Cout, Cin, kh, kw, ph, pw}
    struct LayerDef { int wi, Cout, Cin, kh, kw, ph, pw; };
    const LayerDef defs[15] = {
        { 3, 256, 324, 1, 1, 0, 0},  //  0 convc1
        { 5, 192, 256, 3, 3, 1, 1},  //  1 convc2
        { 7, 128,   2, 7, 7, 3, 3},  //  2 convf1
        { 9,  64, 128, 3, 3, 1, 1},  //  3 convf2
        {11, 126, 256, 3, 3, 1, 1},  //  4 conv
        {13, 128, 384, 1, 5, 0, 2},  //  5 convz1
        {15, 128, 384, 1, 5, 0, 2},  //  6 convr1
        {17, 128, 384, 1, 5, 0, 2},  //  7 convq1
        {19, 128, 384, 5, 1, 2, 0},  //  8 convz2
        {21, 128, 384, 5, 1, 2, 0},  //  9 convr2
        {23, 128, 384, 5, 1, 2, 0},  // 10 convq2
        {25, 256, 128, 3, 3, 1, 1},  // 11 fh1
        {27,   2, 256, 3, 3, 1, 1},  // 12 fh2
        {29, 256, 128, 3, 3, 1, 1},  // 13 mk1
        {31, 576, 256, 1, 1, 0, 0},  // 14 mk2
    };
    _Float16* Wh[15]; float* Bp[15]; int Mp[15], Kp[15], Kr[15];
    for (int l = 0; l < 15; ++l) {
        Kr[l] = defs[l].Cin * defs[l].kh * defs[l].kw;
        Mp[l] = rup(defs[l].Cout, 16);
        Kp[l] = rup(Kr[l], 32);
        Wh[l] = (_Float16*)alloc((size_t)Mp[l] * Kp[l] * 2);
        Bp[l] = (float*)alloc((size_t)Mp[l] * 4);
    }

    _Float16* BT   = (_Float16*)alloc((size_t)PX * 2304 * 2);  // im2col slab (max Kpad)
    float* corrfeat= (float*)alloc((size_t)324 * PX * 4);
    float* A1      = (float*)alloc((size_t)256 * PX * 4);      // co / fh1 / mk1 out
    float* A2      = (float*)alloc((size_t)128 * PX * 4);      // fl
    float* CF      = (float*)alloc((size_t)256 * PX * 4);      // [co2 ; fl2]
    float* hx      = (float*)alloc((size_t)384 * PX * 4);      // [net ; inp ; mf ; flow]
    float* qx      = (float*)alloc((size_t)384 * PX * 4);      // [r*net ; x]
    float* Zs      = (float*)alloc((size_t)128 * PX * 4);
    float* Rs      = (float*)alloc((size_t)128 * PX * 4);
    float* Qs      = (float*)alloc((size_t)128 * PX * 4);
    float* net     = (float*)alloc((size_t)128 * PX * 4);
    float* maskb   = (float*)alloc((size_t)576 * PX * 4);
    float* dfpad   = (float*)alloc((size_t)16  * PX * 4);
    float* flowb   = (float*)alloc((size_t)2   * PX * 4);
    float* cx      = (float*)alloc((size_t)PX * 4);
    float* cy      = (float*)alloc((size_t)PX * 4);
    (void)ws_size; (void)in_sizes; (void)n_in; (void)out_size;

    auto gemm = [&](const _Float16* A, const _Float16* Bt, float* Cf, _Float16* Ch,
                    const float* bias, int Mpad, int Kpad, float alpha, int op) {
        int tiles = (Mpad / 16) * (PX / 64);   // 16x64 strip per wave
        k_gemm_wmma<<<(tiles + 7) / 8, NTHREADS, 0, stream>>>(
            A, Bt, Cf, Ch, bias, Mpad, PX, Kpad, alpha, op, Ch != nullptr ? 1 : 0);
    };
    auto im2col = [&](const float* src, int l) {
        size_t n = (size_t)Kp[l] * PX;
        k_im2col_t<<<nb(n), NTHREADS, 0, stream>>>(BT, src, defs[l].Cin,
            defs[l].kh, defs[l].kw, defs[l].ph, defs[l].pw, Kp[l]);
    };
    auto conv = [&](const float* src, int l, float* Cf, float alpha, int op) {
        im2col(src, l);
        gemm(Wh[l], BT, Cf, nullptr, Bp[l], Mp[l], Kp[l], alpha, op);
    };
    auto copyf = [&](float* d, const float* s, size_t n) {
        k_copy_f32<<<nb(n), NTHREADS, 0, stream>>>(d, s, n);
    };

    // ---- pack weights/biases (f32 -> padded f16) ----
    for (int l = 0; l < 15; ++l) {
        float bscale = (l == 14) ? 0.25f : 1.0f;   // mk = 0.25*(Wx+b)
        size_t nw = (size_t)Mp[l] * Kp[l];
        k_pack_w<<<nb(nw), NTHREADS, 0, stream>>>(Wh[l], (const float*)d_in[defs[l].wi],
                                                  defs[l].Cout, Kr[l], Mp[l], Kp[l]);
        k_pack_b<<<nb(Mp[l]), NTHREADS, 0, stream>>>(Bp[l], (const float*)d_in[defs[l].wi + 1],
                                                     defs[l].Cout, Mp[l], bscale);
    }

    // ---- feature transpose-convert: (256 x 4800) f32 -> (4800 x 256) f16 ----
    k_f32_to_h_t<<<nb((size_t)256 * PX), NTHREADS, 0, stream>>>(f1t, fmap1, 256, PX);
    k_f32_to_h_t<<<nb((size_t)256 * PX), NTHREADS, 0, stream>>>(f2t, fmap2, 256, PX);

    // ---- correlation volume via WMMA: pyr0 = f1t * f2t^T / 16 (f16 out) ----
    {
        int tiles = (PX / 16) * (PX / 64);
        k_gemm_wmma<<<(tiles + 7) / 8, NTHREADS, 0, stream>>>(
            f1t, f2t, nullptr, pyr0, nullptr, PX, PX, 256, 1.0f / 16.0f, 0, 1);
    }
    k_avgpool2<<<nb((size_t)PX * 30 * 40), NTHREADS, 0, stream>>>(pyr1, pyr0, 60, 80);
    k_avgpool2<<<nb((size_t)PX * 15 * 20), NTHREADS, 0, stream>>>(pyr2, pyr1, 30, 40);
    k_avgpool2<<<nb((size_t)PX * 7 * 10),  NTHREADS, 0, stream>>>(pyr3, pyr2, 15, 20);

    // ---- context split: net = tanh(cnet[:128]); hx rows 128..255 = relu(cnet[128:]) ----
    k_split_cnet<<<nb((size_t)256 * PX), NTHREADS, 0, stream>>>(net, hx, cnet);
    k_coords_init<<<nb(PX), NTHREADS, 0, stream>>>(cx, cy);

    // ---- 12 RAFT iterations ----
    for (int it = 0; it < ITERS; ++it) {
        k_flow_from_coords<<<nb(PX), NTHREADS, 0, stream>>>(flowb, cx, cy);
        k_corr_lookup<<<nb((size_t)324 * PX), NTHREADS, 0, stream>>>(
            corrfeat, pyr0, pyr1, pyr2, pyr3, cx, cy);

        conv(corrfeat, 0, A1, 1.0f, 1);                        // co  = relu(convc1)
        conv(A1,       1, CF, 1.0f, 1);                        // co2 -> CF[0:192]
        conv(flowb,    2, A2, 1.0f, 1);                        // fl  = relu(convf1)
        conv(A2,       3, CF + (size_t)192 * PX, 1.0f, 1);     // fl2 -> CF[192:256]
        conv(CF,       4, hx + (size_t)256 * PX, 1.0f, 1);     // mf -> hx[256:384]
        copyf(hx + (size_t)382 * PX, flowb, (size_t)2 * PX);   // flow -> hx[382:384]
        copyf(hx, net, (size_t)128 * PX);                      // net -> hx[0:128]

        // horizontal GRU (1x5)
        im2col(hx, 5);
        gemm(Wh[5], BT, Zs, nullptr, Bp[5], Mp[5], Kp[5], 1.0f, 2); // z
        gemm(Wh[6], BT, Rs, nullptr, Bp[6], Mp[6], Kp[6], 1.0f, 2); // r
        k_mul_f32<<<nb((size_t)128 * PX), NTHREADS, 0, stream>>>(qx, Rs, net, (size_t)128 * PX);
        copyf(qx + (size_t)128 * PX, hx + (size_t)128 * PX, (size_t)256 * PX); // x tail
        conv(qx, 7, Qs, 1.0f, 3);                              // q = tanh(convq1)
        k_gru<<<nb((size_t)128 * PX), NTHREADS, 0, stream>>>(net, Zs, Qs, (size_t)128 * PX);

        // vertical GRU (5x1)
        copyf(hx, net, (size_t)128 * PX);
        im2col(hx, 8);
        gemm(Wh[8], BT, Zs, nullptr, Bp[8], Mp[8], Kp[8], 1.0f, 2);
        gemm(Wh[9], BT, Rs, nullptr, Bp[9], Mp[9], Kp[9], 1.0f, 2);
        k_mul_f32<<<nb((size_t)128 * PX), NTHREADS, 0, stream>>>(qx, Rs, net, (size_t)128 * PX);
        conv(qx, 10, Qs, 1.0f, 3);
        k_gru<<<nb((size_t)128 * PX), NTHREADS, 0, stream>>>(net, Zs, Qs, (size_t)128 * PX);

        // flow head: df = fh2(relu(fh1(net)))
        conv(net, 11, A1, 1.0f, 1);
        conv(A1,  12, dfpad, 1.0f, 0);
        // mask head: mk = 0.25 * mk2(relu(mk1(net)))
        conv(net, 13, A1, 1.0f, 1);
        conv(A1,  14, maskb, 0.25f, 0);

        k_add_coords<<<nb(PX), NTHREADS, 0, stream>>>(cx, cy, dfpad);
        k_flow_from_coords<<<nb(PX), NTHREADS, 0, stream>>>(flowb, cx, cy);
        k_upsample<<<nb((size_t)2 * PX * 64), NTHREADS, 0, stream>>>(
            out + (size_t)it * 2 * 307200, maskb, flowb);
    }
}
